// Attention_40767829574409
// MI455X (gfx1250) — matrix operand order repaired and checked by reading, not compiled, not run on hardware
//
#include <hip/hip_runtime.h>
#include <hip/hip_bf16.h>
#include <stdint.h>

// ---------------------------------------------------------------------------
// Problem constants
// ---------------------------------------------------------------------------
#define NB   32
#define TX   240
#define TY   1200
#define CCH  256     // ENC_HID
#define MEL  80
#define HID  128
#define VOC  100
#define VPAD 112     // vocab padded to 7 WMMA tiles
#define NEGF (-1e9f)

static const size_t A_ELEM = (size_t)NB * CCH * TY;      // 9,830,400
static const size_t S_ELEM = (size_t)NB * TX * TY;       // 9,216,000
static const size_t Q_ELEM = (size_t)NB * TY * HID;      // 4,915,200
static const size_t K_ELEM = (size_t)NB * TX * HID;      //   983,040
static const size_t C_ELEM = (size_t)NB * TY * VOC;      // 3,840,000

typedef __attribute__((ext_vector_type(16))) _Float16 v16h;
typedef __attribute__((ext_vector_type(8)))  _Float16 v8h;
typedef __attribute__((ext_vector_type(8)))  float    v8f;
typedef __attribute__((ext_vector_type(4)))  unsigned u32x4;
typedef __attribute__((ext_vector_type(8)))  int      i32x8;
typedef __attribute__((ext_vector_type(4)))  int      i32x4;

union F16x16 { v16h v; _Float16 e[16]; };
union F32x8  { v8f  v; float     e[8];  };

#if defined(__HIP_DEVICE_COMPILE__) && \
    __has_builtin(__builtin_amdgcn_tensor_load_to_lds) && \
    __has_builtin(__builtin_amdgcn_s_wait_tensorcnt)
#define HAVE_TDM_PATH 1
#endif

__device__ __forceinline__ v16h cat8(v8h lo, v8h hi) {
  return __builtin_shufflevector(lo, hi, 0, 1, 2, 3, 4, 5, 6, 7,
                                 8, 9, 10, 11, 12, 13, 14, 15);
}

// gfx1250 async global->LDS copy (ASYNCcnt). LDS offset is relative to the
// workgroup LDS base; the staging buffer is the kernel's only __shared__
// object, so it sits at offset 0.
__device__ __forceinline__ void async_ld_b32(unsigned lds_off, const void* g) {
  asm volatile("global_load_async_to_lds_b32 %0, %1, off"
               :: "v"(lds_off), "v"((unsigned long long)(uintptr_t)g)
               : "memory");
}
__device__ __forceinline__ void wait_async() {
  asm volatile("s_wait_asynccnt 0x0" ::: "memory");
}

#ifdef HAVE_TDM_PATH
// Tensor Data Mover: DMA a 2D tile [rows x cols f16] (row stride = stride
// elements) from global into LDS offset 0, rows packed at cols*2 bytes.
// D# layout per CDNA5 ISA sec 8.3/8.4 (group0 128b, group1 256b, 2D tensor).
__device__ __forceinline__ void tdm_load_2d_f16(const void* gaddr, unsigned rows,
                                                unsigned cols, unsigned stride,
                                                unsigned tdim0) {
  unsigned long long ga = (unsigned long long)(uintptr_t)gaddr;
  u32x4 g0 = { 1u,                                   // count=1, user mode
               0u,                                   // lds_addr = 0
               (unsigned)(ga & 0xFFFFFFFFu),
               (unsigned)((ga >> 32) & 0x01FFFFFFu) | (2u << 30) };  // type=2
  i32x8 g1 = { (int)(1u << 16),                      // data_size = 2 bytes
               (int)((tdim0 & 0xFFFFu) << 16),       // tensor_dim0[15:0]
               (int)((tdim0 >> 16) | ((rows & 0xFFFFu) << 16)),  // td0 hi|td1 lo
               (int)((rows >> 16) | ((cols & 0xFFFFu) << 16)),   // td1 hi|tile_dim0
               (int)(rows & 0xFFFFu),                // tile_dim1 (tile_dim2=0)
               (int)stride,                          // tensor_dim0_stride lo
               0, 0 };
  i32x4 gz = { 0, 0, 0, 0 };
#if __clang_major__ >= 23
  i32x8 gz8 = { 0, 0, 0, 0, 0, 0, 0, 0 };
  __builtin_amdgcn_tensor_load_to_lds(g0, g1, gz, gz, gz8, 0);
#else
  __builtin_amdgcn_tensor_load_to_lds(g0, g1, gz, gz, 0);
#endif
  __builtin_amdgcn_s_wait_tensorcnt((short)0);
}
#endif

// ---------------------------------------------------------------------------
// Device helpers
// ---------------------------------------------------------------------------
__device__ __forceinline__ float logsigf(float x) {
  return fminf(x, 0.f) - log1pf(expf(-fabsf(x)));
}
__device__ __forceinline__ float laef(float a, float b) {
  float m = fmaxf(a, b), n = fminf(a, b);
  return m + log1pf(expf(n - m));
}
__device__ __forceinline__ unsigned f2ord(float f) {
  unsigned u = __float_as_uint(f);
  return (u & 0x80000000u) ? ~u : (u | 0x80000000u);
}
__device__ __forceinline__ float ord2f(unsigned k) {
  return (k & 0x80000000u) ? __uint_as_float(k & 0x7FFFFFFFu)
                           : __uint_as_float(~k);
}
__device__ __forceinline__ int silf(int x, int tl) {
  int v = (x < tl) ? 1 : 0;
  if (x < TX - 1) v -= ((x + 1) < tl) ? 1 : 0;
  if (x == 0) v = 1;
  return v;
}
__device__ __forceinline__ float bsum(float v, float* sh) {
  int t = threadIdx.x; sh[t] = v; __syncthreads();
  for (int o = blockDim.x >> 1; o > 0; o >>= 1) {
    if (t < o) sh[t] += sh[t + o];
    __syncthreads();
  }
  float r = sh[0]; __syncthreads(); return r;
}
__device__ __forceinline__ float bmaxr(float v, float* sh) {
  int t = threadIdx.x; sh[t] = v; __syncthreads();
  for (int o = blockDim.x >> 1; o > 0; o >>= 1) {
    if (t < o) sh[t] = fmaxf(sh[t], sh[t + o]);
    __syncthreads();
  }
  float r = sh[0]; __syncthreads(); return r;
}
__device__ __forceinline__ float bminr(float v, float* sh) {
  int t = threadIdx.x; sh[t] = v; __syncthreads();
  for (int o = blockDim.x >> 1; o > 0; o >>= 1) {
    if (t < o) sh[t] = fminf(sh[t], sh[t + o]);
    __syncthreads();
  }
  float r = sh[0]; __syncthreads(); return r;
}

// ---------------------------------------------------------------------------
// Fill kernels
// ---------------------------------------------------------------------------
__global__ void k_fillf(float* p, size_t cnt, float v) {
  size_t i = (size_t)blockIdx.x * blockDim.x + threadIdx.x;
  if (i < cnt) p[i] = v;
}
__global__ void k_fillu(unsigned* p, size_t cnt, unsigned v) {
  size_t i = (size_t)blockIdx.x * blockDim.x + threadIdx.x;
  if (i < cnt) p[i] = v;
}

// ---------------------------------------------------------------------------
// Weight pre-pack: f32 [co][ci][3] -> f16 im2col-K-major [co][Kpad], zero pad.
// ---------------------------------------------------------------------------
__global__ void k_prepw(const float* __restrict__ w, _Float16* __restrict__ wA,
                        int Cin) {
  int K = Cin * 3, Kpad = (K + 31) & ~31;
  int total = CCH * Kpad;
  for (int i = blockIdx.x * blockDim.x + threadIdx.x; i < total;
       i += gridDim.x * blockDim.x) {
    int co = i / Kpad, k = i - co * Kpad;
    float v = 0.f;
    if (k < K) { int ci = k / 3, kk = k - 3 * ci; v = w[(co * Cin + ci) * 3 + kk]; }
    wA[i] = (_Float16)v;
  }
}

// Pre-pack f32 [rows_src][CCH] -> f16 [rows_pad][CCH] (zero-padded rows).
__global__ void k_prepw2(const float* __restrict__ src, _Float16* __restrict__ dst,
                         int rows_src, int rows_pad) {
  int total = rows_pad * CCH;
  for (int i = blockIdx.x * blockDim.x + threadIdx.x; i < total;
       i += gridDim.x * blockDim.x) {
    int r = i / CCH, c = i - r * CCH;
    dst[i] = (_Float16)((r < rows_src) ? src[r * CCH + c] : 0.f);
  }
}

// ---------------------------------------------------------------------------
// Conv (k=3, pad 1) as WMMA GEMM with LDS-staged input window.
// One block = (n, 16-wide t tile); 8 waves x 2 c_out tiles = 256 channels.
// Input window raw[ci][20] covers tin in [t0-2, t0+18).  Interior tiles are
// staged with one TDM tensor_load_to_lds (TENSORcnt); edge tiles use per-lane
// async b32 copies (ASYNCcnt); layer 0 converts f32 spec in-shader.
// Masking (conv pad + spec_length) is applied at fragment build.
// f16 fragment layouts per ISA 7.12.2:
//   A lane (m=lane%16, g=lane/16): slots h<8 -> k0+8g+h, h>=8 -> k0+16+8g+h-8
//   B lane (n=lane%16, g):         slots h<8 -> k0+8g+h, h>=8 -> k0+16+8g+h-8
//   C f32  lane (n=lane%16, g):    vgpr v    -> m = 8g+v
// ---------------------------------------------------------------------------
__global__ void k_conv_wmma(const _Float16* __restrict__ actin,
                            const float* __restrict__ spec,
                            const _Float16* __restrict__ wA,
                            const int* __restrict__ sl,
                            float* __restrict__ y, int Cin, int layer0) {
  __shared__ _Float16 raw[CCH * 20];      // only __shared__ object -> LDS off 0
  int blk = blockIdx.x;
  int t_t = blk % 75, n = blk / 75;
  int t0 = t_t * 16;
  int tid = threadIdx.x;
  int K = Cin * 3;
  int Kpad = (K + 31) & ~31;
  int slen = sl[n];

  // zero the whole window (covers pad columns + channel rows beyond Cin)
  for (int i = tid; i < CCH * 10; i += 256) ((unsigned*)raw)[i] = 0u;
  __syncthreads();

  if (layer0) {
    for (int i = tid; i < Cin * 20; i += 256) {
      int ci = i / 20, j = i - ci * 20;
      int tin = t0 - 2 + j;
      if (tin >= 0 && tin < TY)
        raw[ci * 20 + j] = (_Float16)spec[((size_t)n * TY + tin) * MEL + ci];
    }
  }
#ifdef HAVE_TDM_PATH
  else if (t0 >= 2 && t0 + 18 <= TY) {
    // interior tile: rows pack at exactly 20 halves -> matches raw[ci][20]
    if (tid < 32)
      tdm_load_2d_f16(&actin[((size_t)n * CCH) * TY + (t0 - 2)],
                      (unsigned)Cin, 20u, (unsigned)TY,
                      (unsigned)(TY - (t0 - 2)));
  }
#endif
  else {
    for (int i = tid; i < Cin * 10; i += 256) {
      int ci = i / 10, jd = i - ci * 10;
      int tin = t0 - 2 + jd * 2;
      if (tin >= 0 && tin + 1 < TY) {
        unsigned lds_off = (unsigned)(ci * 20 + jd * 2) * 2u;
        async_ld_b32(lds_off, &actin[((size_t)n * CCH + ci) * TY + tin]);
      }
    }
    wait_async();
  }
  __syncthreads();

  int wv = tid >> 5, lane = tid & 31;
  int g = lane >> 4, lm = lane & 15;
  int cot0 = wv * 2, cot1 = wv * 2 + 1;
  int coA0 = cot0 * 16 + lm, coA1 = cot1 * 16 + lm;

  F32x8 acc0, acc1;
#pragma unroll
  for (int i = 0; i < 8; i++) { acc0.e[i] = 0.f; acc1.e[i] = 0.f; }

#pragma unroll 2
  for (int k0 = 0; k0 < K; k0 += 32) {
    F16x16 a0, a1, b;
    a0.v = cat8(*(const v8h*)&wA[(size_t)coA0 * Kpad + k0 + 8 * g],
                *(const v8h*)&wA[(size_t)coA0 * Kpad + k0 + 16 + 8 * g]);
    a1.v = cat8(*(const v8h*)&wA[(size_t)coA1 * Kpad + k0 + 8 * g],
                *(const v8h*)&wA[(size_t)coA1 * Kpad + k0 + 16 + 8 * g]);
#pragma unroll
    for (int h = 0; h < 16; h++) {
      int k = k0 + ((h < 8) ? h : h + 8) + 8 * g;
      int ci = k / 3, kk = k - 3 * ci;
      int tin = t0 + lm + kk - 1;
      _Float16 xv = raw[ci * 20 + lm + kk + 1];
      b.e[h] = (tin < slen) ? xv : (_Float16)0.f;
    }
    if (k0 + 32 < K)
      __builtin_prefetch(&wA[(size_t)coA0 * Kpad + k0 + 32], 0, 1);
    acc0.v = __builtin_amdgcn_wmma_f32_16x16x32_f16(false, a0.v, false, b.v,
                                                    (short)0, acc0.v, false, false);
    acc1.v = __builtin_amdgcn_wmma_f32_16x16x32_f16(false, a1.v, false, b.v,
                                                    (short)0, acc1.v, false, false);
  }
  int t = t0 + lm;
#pragma unroll
  for (int v = 0; v < 8; v++) {
    y[((size_t)n * CCH + cot0 * 16 + g * 8 + v) * TY + t] = acc0.e[v];
    y[((size_t)n * CCH + cot1 * 16 + g * 8 + v) * TY + t] = acc1.e[v];
  }
}

// Per-channel BatchNorm statistics (biased variance, torch behavior).
__global__ void k_bn_stats(const float* __restrict__ y, float* meanb, float* rstdb) {
  int c = blockIdx.x;
  float s = 0.f, s2 = 0.f;
  for (int i = threadIdx.x; i < NB * TY; i += blockDim.x) {
    int n = i / TY, t = i - n * TY;
    float v = y[((size_t)n * CCH + c) * TY + t];
    s += v; s2 += v * v;
  }
  __shared__ float sh[256];
  float ts = bsum(s, sh);
  float ts2 = bsum(s2, sh);
  if (threadIdx.x == 0) {
    float m = ts / (float)(NB * TY);
    float var = ts2 / (float)(NB * TY) - m * m;
    meanb[c] = m;
    rstdb[c] = rsqrtf(fmaxf(var, 0.f) + 1e-5f);
  }
}

// BN apply + optional residual add/set + optional relu; writes f16 activation.
__global__ void k_bn_apply(const float* __restrict__ y,
                           const float* __restrict__ meanb,
                           const float* __restrict__ rstdb,
                           const float* __restrict__ gma,
                           const float* __restrict__ bta,
                           const _Float16* __restrict__ res_in,
                           _Float16* __restrict__ res_out,
                           _Float16* __restrict__ actout,
                           int addres, int setres, int dorelu) {
  size_t i = (size_t)blockIdx.x * blockDim.x + threadIdx.x;
  if (i >= A_ELEM) return;
  int c = (int)((i / TY) % CCH);
  float v = (y[i] - meanb[c]) * rstdb[c] * gma[c] + bta[c];
  if (addres) v += (float)res_in[i];
  if (setres) res_out[i] = (_Float16)v;
  if (dorelu) v = fmaxf(v, 0.f);
  actout[i] = (_Float16)v;
}

// ---------------------------------------------------------------------------
// keyn: embed, mask, L2-normalize -> f16
// ---------------------------------------------------------------------------
__global__ void k_keyn(const int* __restrict__ text, const float* __restrict__ emb,
                       const float* __restrict__ tmask, _Float16* __restrict__ kn) {
  int row = blockIdx.x;               // n*TX + x
  int h = threadIdx.x;                // 128
  float v = emb[text[row] * HID + h] * tmask[row];
  __shared__ float sh[HID];
  float nrm2 = bsum(v * v, sh);
  float nrm = fmaxf(sqrtf(nrm2), 1e-8f);
  kn[(size_t)row * HID + h] = (_Float16)(v / nrm);
}

// ---------------------------------------------------------------------------
// query = enc0^T @ q_w^T + q_b  (M=y tile, N=h tile, K=256) via WMMA
// ---------------------------------------------------------------------------
__global__ void k_query_wmma(const _Float16* __restrict__ enc0,
                             const _Float16* __restrict__ qwh,
                             const float* __restrict__ qb,
                             float* __restrict__ qf32) {
  int wid  = blockIdx.x * (blockDim.x >> 5) + (threadIdx.x >> 5);
  int lane = threadIdx.x & 31;
  int h_t = wid % 8; int tmp = wid / 8;
  int y_t = tmp % 75; int n = tmp / 75;
  if (n >= NB) return;
  int g = lane >> 4, lm = lane & 15;
  int hh = h_t * 16 + lm;

  F32x8 acc;
#pragma unroll
  for (int i = 0; i < 8; i++) acc.e[i] = 0.f;

#pragma unroll
  for (int k0 = 0; k0 < CCH; k0 += 32) {
    F16x16 a, b;
#pragma unroll
    for (int h = 0; h < 16; h++) {
      int c = h + 16 * g + k0;
      a.e[h] = enc0[((size_t)n * CCH + c) * TY + y_t * 16 + lm];
    }
    b.v = cat8(*(const v8h*)&qwh[(size_t)hh * CCH + k0 + 16 * g],
               *(const v8h*)&qwh[(size_t)hh * CCH + k0 + 16 * g + 8]);
    acc.v = __builtin_amdgcn_wmma_f32_16x16x32_f16(false, a.v, false, b.v,
                                                   (short)0, acc.v, false, false);
  }
#pragma unroll
  for (int v = 0; v < 8; v++) {
    int y = y_t * 16 + g * 8 + v;
    qf32[((size_t)n * TY + y) * HID + hh] = acc.e[v] + qb[hh];
  }
}

__global__ void k_normq(const float* __restrict__ qf32, _Float16* __restrict__ qn) {
  int row = blockIdx.x;               // n*TY + y
  int h = threadIdx.x;
  float v = qf32[(size_t)row * HID + h];
  __shared__ float sh[HID];
  float nrm2 = bsum(v * v, sh);
  float nrm = fmaxf(sqrtf(nrm2), 1e-8f);
  qn[(size_t)row * HID + h] = (_Float16)(v / nrm);
}

// ---------------------------------------------------------------------------
// cos = keyn @ queryn^T  (M=x, N=y, K=128) -> similarity
// ---------------------------------------------------------------------------
__global__ void k_cos_wmma(const _Float16* __restrict__ kn,
                           const _Float16* __restrict__ qn,
                           const float* __restrict__ stm,
                           const float* __restrict__ sim_w,
                           const float* __restrict__ sim_b,
                           float* __restrict__ sim) {
  int wid  = blockIdx.x * (blockDim.x >> 5) + (threadIdx.x >> 5);
  int lane = threadIdx.x & 31;
  int y_t = wid % 75; int tmp = wid / 75;
  int x_t = tmp % 15; int n = tmp / 15;
  if (n >= NB) return;
  int g = lane >> 4, lm = lane & 15;
  const _Float16* arow = &kn[((size_t)n * TX + x_t * 16 + lm) * HID];
  const _Float16* brow = &qn[((size_t)n * TY + y_t * 16 + lm) * HID];

  F32x8 acc;
#pragma unroll
  for (int i = 0; i < 8; i++) acc.e[i] = 0.f;

#pragma unroll
  for (int k0 = 0; k0 < HID; k0 += 32) {
    F16x16 a, b;
    a.v = cat8(*(const v8h*)&arow[k0 + 8 * g],
               *(const v8h*)&arow[k0 + 16 + 8 * g]);
    b.v = cat8(*(const v8h*)&brow[k0 + 8 * g],
               *(const v8h*)&brow[k0 + 16 + 8 * g]);
    acc.v = __builtin_amdgcn_wmma_f32_16x16x32_f16(false, a.v, false, b.v,
                                                   (short)0, acc.v, false, false);
  }
  float scale = 10.f * expf(sim_w[0]);
  float sb = sim_b[0];
#pragma unroll
  for (int v = 0; v < 8; v++) {
    int x = x_t * 16 + g * 8 + v;
    int y = y_t * 16 + lm;
    float st = stm[n * TX + x];
    float c = (1.f - st) * acc.e[v] - st;
    sim[((size_t)n * TX + x) * TY + y] = scale * c + sb;
  }
}

// ---------------------------------------------------------------------------
// ctc_logit = enc1^T @ ctc_w^T + ctc_b (M=y, N=v padded to 112, K=256)
// ---------------------------------------------------------------------------
__global__ void k_ctcgemm_wmma(const _Float16* __restrict__ enc1,
                               const _Float16* __restrict__ cwh,
                               const float* __restrict__ cb,
                               float* __restrict__ logit) {
  int wid  = blockIdx.x * (blockDim.x >> 5) + (threadIdx.x >> 5);
  int lane = threadIdx.x & 31;
  int v_t = wid % 7; int tmp = wid / 7;
  int y_t = tmp % 75; int n = tmp / 75;
  if (n >= NB) return;
  int g = lane >> 4, lm = lane & 15;
  int vc = v_t * 16 + lm;

  F32x8 acc;
#pragma unroll
  for (int i = 0; i < 8; i++) acc.e[i] = 0.f;

#pragma unroll
  for (int k0 = 0; k0 < CCH; k0 += 32) {
    F16x16 a, b;
#pragma unroll
    for (int h = 0; h < 16; h++) {
      int c = h + 16 * g + k0;
      a.e[h] = enc1[((size_t)n * CCH + c) * TY + y_t * 16 + lm];
    }
    b.v = cat8(*(const v8h*)&cwh[(size_t)vc * CCH + k0 + 16 * g],
               *(const v8h*)&cwh[(size_t)vc * CCH + k0 + 16 * g + 8]);
    acc.v = __builtin_amdgcn_wmma_f32_16x16x32_f16(false, a.v, false, b.v,
                                                   (short)0, acc.v, false, false);
  }
  if (vc < VOC) {
#pragma unroll
    for (int v = 0; v < 8; v++) {
      int y = y_t * 16 + g * 8 + v;
      logit[((size_t)n * TY + y) * VOC + vc] = acc.e[v] + cb[vc];
    }
  }
}

// logsumexp over vocab + spectral energy per (n, y)
__global__ void k_logz_energy(const float* __restrict__ logit,
                              const float* __restrict__ spec,
                              float* __restrict__ logz,
                              float* __restrict__ energy) {
  int row = blockIdx.x;               // n*TY + y
  int t = threadIdx.x;                // 128
  float v = (t < VOC) ? logit[(size_t)row * VOC + t] : NEGF;
  __shared__ float sh[128];
  float mx = bmaxr(v, sh);
  float ev = (t < VOC) ? expf(v - mx) : 0.f;
  float s = bsum(ev, sh);
  if (t == 0) logz[row] = mx + logf(s);
  float e = (t < 60) ? expf(spec[(size_t)row * MEL + 20 + t]) : 0.f;
  float es = bsum(e, sh);
  if (t == 0) energy[row] = es / 60.f;
}

// ---------------------------------------------------------------------------
// Global minima of lsmx / lsmx2 (ordered-uint atomicMin)
// ---------------------------------------------------------------------------
__global__ void k_min_sim(const float* __restrict__ sim, const int* tl, const int* sl,
                          unsigned* minv) {
  int row = blockIdx.x; int n = row / TX, x = row - n * TX;
  int tln = tl[n], sln = sl[n];
  float mn = 3.0e38f;
  for (int y = threadIdx.x; y < TY; y += blockDim.x) {
    float v = (x < tln && y < sln) ? logsigf(sim[(size_t)row * TY + y]) : 0.f;
    mn = fminf(mn, v);
  }
  __shared__ float sh[256];
  float r = bminr(mn, sh);
  if (threadIdx.x == 0) atomicMin(&minv[0], f2ord(r));
}

__global__ void k_min_ctc(const float* __restrict__ logit, const float* __restrict__ logz,
                          const int* __restrict__ text, const int* tl, const int* sl,
                          unsigned* minv) {
  int row = blockIdx.x; int n = row / TX, x = row - n * TX;
  int tln = tl[n], sln = sl[n];
  int tok = text[row];
  float mn = 3.0e38f;
  for (int y = threadIdx.x; y < TY; y += blockDim.x) {
    float v = 0.f;
    if (x < tln && y < sln) {
      float p = expf(logit[((size_t)n * TY + y) * VOC + tok] - logz[n * TY + y]);
      v = logsigf(p);
    }
    mn = fminf(mn, v);
  }
  __shared__ float sh[256];
  float r = bminr(mn, sh);
  if (threadIdx.x == 0) atomicMin(&minv[1], f2ord(r));
}

// ---------------------------------------------------------------------------
// Monotonic alignment search: forward DP (one block per batch row)
// ---------------------------------------------------------------------------
__global__ void k_mas_fwd(const float* __restrict__ sim,
                          const float* __restrict__ logit,
                          const float* __restrict__ logz,
                          const int* __restrict__ text,
                          const int* tl, const int* sl,
                          const unsigned* minv, float* __restrict__ dp, int mode) {
  int n = blockIdx.x;
  int x = threadIdx.x;                // 256 threads, TX=240 active
  __shared__ float cur[256], nxt[256];
  float minval = ord2f(minv[mode]);
  int tln = tl[n], sln = sl[n];
  cur[x] = NEGF;
  __syncthreads();
  for (int y = 0; y < TY; y++) {
    if (x < TX) {
      bool m = (x < tln) && (y < sln);
      float ls;
      if (mode == 0) {
        ls = logsigf(sim[((size_t)n * TX + x) * TY + y]);
      } else {
        float p = expf(logit[((size_t)n * TY + y) * VOC + text[n * TX + x]] -
                       logz[n * TY + y]);
        ls = logsigf(p);
      }
      float lsm = m ? ls : 0.f;
      float val = (lsm == 0.f) ? (lsm - minval) : lsm;
      float dpx = cur[x];
      float dpxm = (x > 0) ? cur[x - 1] : NEGF;
      float best = fmaxf(dpx, dpxm);
      float base = (y == 0) ? ((x == 0) ? 0.f : NEGF) : best;
      float dnew = val + base;
      bool valid = (x <= y) && (x >= tln + y - sln) && (x < tln) && (y < sln);
      dnew = valid ? dnew : NEGF;
      nxt[x] = dnew;
      dp[((size_t)n * TX + x) * TY + y] = dnew;
    }
    __syncthreads();
    if (x < TX) cur[x] = nxt[x];
    __syncthreads();
  }
}

// Backtrack (sequential per batch row)
__global__ void k_mas_bt(const float* __restrict__ dp, const int* tl, const int* sl,
                         float* __restrict__ path, int* __restrict__ xpath, int mode) {
  int n = blockIdx.x;
  if (threadIdx.x != 0) return;
  int idx = tl[n] - 1;
  int sln = sl[n];
  for (int y = TY - 1; y >= 0; y--) {
    if (y < sln) {
      path[((size_t)n * TX + idx) * TY + y] = 1.f;
      if (mode == 0) xpath[n * TY + y] = idx;
      int ym1 = (y - 1 > 0) ? (y - 1) : 0;
      float vc = dp[((size_t)n * TX + idx) * TY + ym1];
      int im1 = (idx > 0) ? (idx - 1) : 0;
      float vp = dp[((size_t)n * TX + im1) * TY + ym1];
      bool move = (idx > 0) && (y > 0) && ((idx == y) || (vc < vp));
      if (move) idx -= 1;
    }
  }
}

// ---------------------------------------------------------------------------
// Silence promotion threshold db per batch row
// ---------------------------------------------------------------------------
__global__ void k_sil_db(const float* __restrict__ energy, const int* __restrict__ xpath,
                         const int* tl, const int* sl, float* __restrict__ db) {
  int n = blockIdx.x;
  int tln = tl[n], sln = sl[n];
  float mx = 0.f, mn = 3.0e38f;
  for (int y = threadIdx.x; y < TY; y += blockDim.x) {
    float se, nsv;
    if (y < sln) {
      int x = xpath[n * TY + y];
      int s = silf(x, tln);
      float e = energy[n * TY + y];
      se = e * (float)s;
      float ns = e * (float)(1 - s);
      nsv = (ns == 0.f) ? 100.f : ns;
    } else {
      se = 0.f; nsv = 100.f;
    }
    mx = fmaxf(mx, se); mn = fminf(mn, nsv);
  }
  __shared__ float sh[256];
  float rmx = bmaxr(mx, sh);
  float rmn = bminr(mn, sh);
  if (threadIdx.x == 0) db[n] = 0.5f * (rmx + rmn);
}

// ---------------------------------------------------------------------------
// Fused loss reduction: inter, aux (3-tap smoothed path), denom, promo sums
// acc layout: [0,32) inter | [32,64) aux | [64,96) denom | [96,128) pnum |
//             [128,160) pden | [160] ctc_sum
// ---------------------------------------------------------------------------
__global__ void k_loss(const float* __restrict__ sim, const float* __restrict__ att,
                       const float* __restrict__ paux, const float* __restrict__ energy,
                       const float* __restrict__ db, const int* tl, const int* sl,
                       float* __restrict__ acc) {
  int row = blockIdx.x; int n = row / TX, x = row - n * TX;
  int tln = tl[n], sln = sl[n];
  float sil = (float)silf(x, tln);
  float dbn = db[n];
  float s_inter = 0.f, s_aux = 0.f, s_den = 0.f, s_pn = 0.f, s_pd = 0.f;
  for (int y = threadIdx.x; y < TY; y += blockDim.x) {
    float m = (x < tln && y < sln) ? 1.f : 0.f;
    float sv = sim[(size_t)row * TY + y];
    float ls = logsigf(sv);
    float nls = logsigf(-sv);
    float mt = att[(size_t)row * TY + y];
    s_inter += -(mt * ls + (1.f - mt) * m * nls);
    float aa = paux[(size_t)row * TY + y];
    if (x > 0)      aa += paux[((size_t)row - 1) * TY + y];
    if (x < TX - 1) aa += paux[((size_t)row + 1) * TY + y];
    aa *= m;
    s_aux += -(aa * ls + (1.f - aa) * m * nls);
    s_den += m;
    float e = energy[n * TY + y];
    float p = ((e <= dbn) ? 1.f : 0.f) * sil;
    s_pn += p * m * ls;
    s_pd += p * m;
  }
  __shared__ float sh[256];
  float r;
  r = bsum(s_inter, sh); if (threadIdx.x == 0) atomicAdd(&acc[n], r);
  r = bsum(s_aux, sh);   if (threadIdx.x == 0) atomicAdd(&acc[32 + n], r);
  r = bsum(s_den, sh);   if (threadIdx.x == 0) atomicAdd(&acc[64 + n], r);
  r = bsum(s_pn, sh);    if (threadIdx.x == 0) atomicAdd(&acc[96 + n], r);
  r = bsum(s_pd, sh);    if (threadIdx.x == 0) atomicAdd(&acc[128 + n], r);
}

// ---------------------------------------------------------------------------
// CTC forward (one block per batch row, L = 2*TX+1 = 481 states in LDS)
// ---------------------------------------------------------------------------
__global__ void k_ctc(const float* __restrict__ logit, const float* __restrict__ logz,
                      const int* __restrict__ text, const int* tl, const int* sl,
                      float* __restrict__ acc) {
  const int L = 2 * TX + 1;
  int n = blockIdx.x;
  int s = threadIdx.x;                // 512 threads
  __shared__ float a0[L], a1[L];
  int zs = 0; bool skip = false;
  if (s < L) {
    zs = (s & 1) ? text[n * TX + (s >> 1)] : 0;
    int zm2 = (s >= 2) ? ((s & 1) ? text[n * TX + ((s - 2) >> 1)] : 0) : 0;
    skip = (s & 1) && (s >= 2) && (zs != zm2);
  }
  if (s < L) {
    float e = logit[((size_t)n * TY + 0) * VOC + zs] - logz[n * TY + 0];
    a0[s] = (s < 2) ? e : NEGF;
  }
  __syncthreads();
  int sln = sl[n];
  float* cur = a0; float* nxt = a1;
  for (int t = 1; t < TY; t++) {
    if (s < L) {
      float al = cur[s];
      float a2 = (s >= 1) ? cur[s - 1] : NEGF;
      float a3 = (skip && s >= 2) ? cur[s - 2] : NEGF;
      float e = logit[((size_t)n * TY + t) * VOC + zs] - logz[n * TY + t];
      float nv = e + laef(laef(al, a2), a3);
      nxt[s] = (t < sln) ? nv : al;
    }
    __syncthreads();
    float* tp = cur; cur = nxt; nxt = tp;
  }
  if (s == 0) {
    int tn = tl[n];
    float ll = laef(cur[2 * tn], cur[2 * tn - 1]);
    atomicAdd(&acc[160], (-ll) / (float)tn);
  }
}

// ---------------------------------------------------------------------------
// Final scalars + att_mask writer
// ---------------------------------------------------------------------------
__global__ void k_final(const float* __restrict__ acc, float* __restrict__ out) {
  if (threadIdx.x != 0 || blockIdx.x != 0) return;
  float nll = 0.f, am = 0.f, sp = 0.f;
  for (int n = 0; n < NB; n++) {
    float den = acc[64 + n];
    nll += acc[n] / den;
    am  += acc[32 + n] / den;
    sp  += -0.01f * acc[96 + n] / fmaxf(acc[128 + n], 1.f);
  }
  nll *= (1.f / NB); am *= (1.f / NB); sp *= (1.f / NB);
  float ctc = acc[160] * (1.f / NB);
  out[(size_t)NB * TX * TY] = nll + 0.5f * am + ctc + sp;   // att_loss
  out[2ull * NB * TX * TY + 1] = nll;                        // nll
}

__global__ void k_attmask(const int* tl, const int* sl, float* __restrict__ out) {
  size_t i = (size_t)blockIdx.x * blockDim.x + threadIdx.x;
  if (i >= (size_t)NB * TX * TY) return;
  int y = (int)(i % TY); size_t r = i / TY;
  int x = (int)(r % TX); int n = (int)(r / TX);
  out[(size_t)NB * TX * TY + 1 + i] = (x < tl[n] && y < sl[n]) ? 1.f : 0.f;
}

// ---------------------------------------------------------------------------
// Host launcher
// ---------------------------------------------------------------------------
#define CDIV(a, b) (((a) + (b) - 1) / (b))

extern "C" void kernel_launch(void* const* d_in, const int* in_sizes, int n_in,
                              void* d_out, int out_size, void* d_ws, size_t ws_size,
                              hipStream_t stream) {
  (void)in_sizes; (void)n_in; (void)out_size; (void)ws_size;

  const int*   text  = (const int*)d_in[0];
  const float* spec  = (const float*)d_in[1];
  const int*   tl    = (const int*)d_in[2];
  const int*   sl    = (const int*)d_in[3];
  const float* tmask = (const float*)d_in[4];
  const float* stm   = (const float*)d_in[5];
  const float* emb   = (const float*)d_in[6];
  const float* conv_w[2][6]; const float* bn_g[2][6]; const float* bn_b[2][6];
  for (int i = 0; i < 6; i++) {
    conv_w[0][i] = (const float*)d_in[7 + i];
    bn_g[0][i]   = (const float*)d_in[13 + i];
    bn_b[0][i]   = (const float*)d_in[19 + i];
    conv_w[1][i] = (const float*)d_in[25 + i];
    bn_g[1][i]   = (const float*)d_in[31 + i];
    bn_b[1][i]   = (const float*)d_in[37 + i];
  }
  const float* qw  = (const float*)d_in[43];
  const float* qb  = (const float*)d_in[44];
  const float* cw  = (const float*)d_in[45];
  const float* cb  = (const float*)d_in[46];
  const float* smw = (const float*)d_in[47];
  const float* smb = (const float*)d_in[48];

  float* out = (float*)d_out;

  // Workspace layout (float units)
  float* W = (float*)d_ws;
  size_t o = 0;
  float*     y_buf = W + o; o += A_ELEM;
  _Float16*  act0  = (_Float16*)(W + o); o += A_ELEM / 2;
  _Float16*  act1  = (_Float16*)(W + o); o += A_ELEM / 2;
  _Float16*  resb  = (_Float16*)(W + o); o += A_ELEM / 2;
  _Float16*  enc0  = (_Float16*)(W + o); o += A_ELEM / 2;
  _Float16*  enc1  = (_Float16*)(W + o); o += A_ELEM / 2;
  float*     sim   = W + o; o += S_ELEM;
  float*     dp    = W + o; o += S_ELEM;
  float*     paux  = W + o; o += S_ELEM;
  float*     qf32  = W + o; o += Q_ELEM;
  _Float16*  qn    = (_Float16*)(W + o); o += Q_ELEM / 2;
  _Float16*  kn    = (_Float16*)(W + o); o += K_ELEM / 2;
  float*     logit = W + o; o += C_ELEM;
  float*     logz  = W + o; o += (size_t)NB * TY;
  float*     energy= W + o; o += (size_t)NB * TY;
  float*     meanb = W + o; o += CCH;
  float*     rstdb = W + o; o += CCH;
  unsigned*  minv  = (unsigned*)(W + o); o += 2;
  int*       xpath = (int*)(W + o); o += (size_t)NB * TY;
  float*     db    = W + o; o += NB;
  float*     acc   = W + o; o += 192;
  _Float16*  wAbuf = (_Float16*)(W + o); o += (size_t)CCH * 768 / 2;
  _Float16*  qwh   = (_Float16*)(W + o); o += (size_t)HID * CCH / 2;
  _Float16*  cwh   = (_Float16*)(W + o); o += (size_t)VPAD * CCH / 2;

  // -- init scratch / output regions that are accumulated into --
  k_fillf<<<(unsigned)CDIV(S_ELEM, 256), 256, 0, stream>>>(out, S_ELEM, 0.f);
  k_fillf<<<(unsigned)CDIV(S_ELEM, 256), 256, 0, stream>>>(paux, S_ELEM, 0.f);
  k_fillf<<<1, 256, 0, stream>>>(acc, 192, 0.f);
  k_fillu<<<1, 32, 0, stream>>>(minv, 2, 0xFFFFFFFFu);

  // -- two conv stacks (weight pre-pack + WMMA GEMM + BN stats + BN apply) --
  _Float16* const layer_in[6]  = {nullptr, act0, act1, act0, act1, act0};
  const int addres[6] = {0, 0, 0, 1, 0, 1};
  const int setres[6] = {0, 1, 0, 1, 0, 0};
  const int dorelu[6] = {1, 1, 1, 1, 1, 0};
  for (int st = 0; st < 2; st++) {
    _Float16* encout = st ? enc1 : enc0;
    _Float16* const layer_out[6] = {act0, act1, act0, act1, act0, encout};
    for (int i = 0; i < 6; i++) {
      int Cin = i ? CCH : MEL;
      k_prepw<<<768, 256, 0, stream>>>(conv_w[st][i], wAbuf, Cin);
      k_conv_wmma<<<NB * 75, 256, 0, stream>>>(layer_in[i], spec, wAbuf, sl,
                                               y_buf, Cin, (i == 0) ? 1 : 0);
      k_bn_stats<<<CCH, 256, 0, stream>>>(y_buf, meanb, rstdb);
      k_bn_apply<<<(unsigned)CDIV(A_ELEM, 256), 256, 0, stream>>>(
          y_buf, meanb, rstdb, bn_g[st][i], bn_b[st][i],
          resb, resb, layer_out[i], addres[i], setres[i], dorelu[i]);
    }
  }

  // -- key / query / cosine similarity --
  k_prepw2<<<128, 256, 0, stream>>>(qw, qwh, HID, HID);
  k_prepw2<<<128, 256, 0, stream>>>(cw, cwh, VOC, VPAD);
  k_keyn<<<NB * TX, HID, 0, stream>>>(text, emb, tmask, kn);
  k_query_wmma<<<2400, 256, 0, stream>>>(enc0, qwh, qb, qf32);
  k_normq<<<NB * TY, HID, 0, stream>>>(qf32, qn);
  k_cos_wmma<<<4500, 256, 0, stream>>>(kn, qn, stm, smw, smb, sim);

  // -- ctc logits + logZ + energy --
  k_ctcgemm_wmma<<<2100, 256, 0, stream>>>(enc1, cwh, cb, logit);
  k_logz_energy<<<NB * TY, 128, 0, stream>>>(logit, spec, logz, energy);

  // -- global minima for lsmx adjustments --
  k_min_sim<<<NB * TX, 256, 0, stream>>>(sim, tl, sl, minv);
  k_min_ctc<<<NB * TX, 256, 0, stream>>>(logit, logz, text, tl, sl, minv);

  // -- monotonic alignment search (primary -> attention, aux -> paux) --
  k_mas_fwd<<<NB, 256, 0, stream>>>(sim, logit, logz, text, tl, sl, minv, dp, 0);
  k_mas_bt<<<NB, 32, 0, stream>>>(dp, tl, sl, out, xpath, 0);
  k_mas_fwd<<<NB, 256, 0, stream>>>(sim, logit, logz, text, tl, sl, minv, dp, 1);
  k_mas_bt<<<NB, 32, 0, stream>>>(dp, tl, sl, paux, xpath, 1);

  // -- silence threshold, fused losses, CTC, finals --
  k_sil_db<<<NB, 256, 0, stream>>>(energy, xpath, tl, sl, db);
  k_loss<<<NB * TX, 256, 0, stream>>>(sim, out, paux, energy, db, tl, sl, acc);
  k_ctc<<<NB, 512, 0, stream>>>(logit, logz, text, tl, sl, acc);
  k_final<<<1, 32, 0, stream>>>(acc, out);
  k_attmask<<<(unsigned)CDIV(S_ELEM, 256), 256, 0, stream>>>(tl, sl, out);
}
